// PrunableLinear_40286793236982
// MI455X (gfx1250) — matrix operand order repaired and checked by reading, not compiled, not run on hardware
//
#include <hip/hip_runtime.h>

typedef __attribute__((ext_vector_type(16))) _Float16 v16h;
typedef __attribute__((ext_vector_type(8)))  _Float16 v8h;
typedef __attribute__((ext_vector_type(4)))  _Float16 v4h;
typedef __attribute__((ext_vector_type(8)))  float    v8f;
typedef __attribute__((ext_vector_type(4)))  float    v4f;

#define M_TOTAL   8192          // Bsz*S = 4*2048
#define IN_DIM    4096
#define OUT_DIM   4096          // R*16
#define R_DIM     256
#define K_DIM     128
#define M_TILE    32            // rows staged in LDS per workgroup
#define LDS_ROW   4104          // 4096 + 8 pad halves -> row stride 8208B == 16B mod 256B (bank-conflict free)
#define THREADS   512           // 16 waves: 2 row-groups x 8 r-values per pass

// ---------------------------------------------------------------------------
// Kernel 1: convert W (R,K,16,16) fp32 -> f16 into workspace (same layout).
// 8,388,608 elements = 2,097,152 float4 quads.
// ---------------------------------------------------------------------------
__global__ __launch_bounds__(256) void wcvt_kernel(const v4f* __restrict__ w,
                                                   v4h* __restrict__ wh) {
  const int q = blockIdx.x * 256 + threadIdx.x;
  v4f v = w[q];
  v4h h = {(_Float16)v.x, (_Float16)v.y, (_Float16)v.z, (_Float16)v.w};
  wh[q] = h;
}

// ---------------------------------------------------------------------------
// Kernel 2: block-sparse GEMM via v_wmma_f32_16x16x32_f16.
// grid = M_TOTAL/M_TILE = 256 workgroups of 512 threads.
// ---------------------------------------------------------------------------
__global__ __launch_bounds__(THREADS) void bsl_kernel(
    const float*    __restrict__ x,     // (8192, 4096) fp32
    const _Float16* __restrict__ wh,    // (256, 128, 16, 16) f16
    const int*      __restrict__ cidx,  // (256, 128)
    float*          __restrict__ out)   // (8192, 4096) fp32
{
  __shared__ _Float16 xs[M_TILE * LDS_ROW];   // 262,656 bytes

  const int tid      = threadIdx.x;
  const int row_base = blockIdx.x * M_TILE;

  // ---- Stage x tile: 32 rows x 4096 fp32 -> f16 in LDS (read x from HBM once)
  // 32*1024 = 32768 float4 quads, 512 threads -> 64 iterations, coalesced.
  const v4f* xq = (const v4f*)x;
#pragma unroll 4
  for (int it = 0; it < (M_TILE * (IN_DIM / 4)) / THREADS; ++it) {
    const int q    = it * THREADS + tid;
    const int row  = q >> 10;          // 1024 quads per row
    const int col4 = q & 1023;
    v4f v = xq[(size_t)(row_base + row) * (IN_DIM / 4) + col4];
    v4h h = {(_Float16)v.x, (_Float16)v.y, (_Float16)v.z, (_Float16)v.w};
    *(v4h*)&xs[row * LDS_ROW + col4 * 4] = h;
  }
  __syncthreads();

  // ---- Wave / lane mapping
  const int wave   = __builtin_amdgcn_readfirstlane(tid >> 5); // 0..15, scalar
  const int lane   = tid & 31;
  const int nm     = lane & 15;       // A row m == B/C column n
  const int half   = lane >> 4;       // lane half
  const int mgroup = wave >> 3;       // 0..1 : which 16-row group
  const int rsub   = wave & 7;        // 0..7 : r within pass

  // A-operand base: row (mgroup*16 + m), K-offset 8*half inside each 16-block
  const _Float16* xa = xs + (mgroup * 16 + nm) * LDS_ROW + 8 * half;

  // ---- Loop over all 256 r blocks (8 per pass)
  for (int rc = 0; rc < R_DIM / 8; ++rc) {
    const int r = rc * 8 + rsub;
    const int*      ci = cidx + r * K_DIM;                  // scalar (r uniform)
    const _Float16* wr = wh + (size_t)r * (K_DIM * 256);

    v8f acc0 = {};
    v8f acc1 = {};

#pragma unroll 2
    for (int p = 0; p < K_DIM / 2; p += 2) {
      // ---- pair p: k-blocks (2p, 2p+1) -> acc0
      {
        const int c0 = ci[2 * p];
        const int c1 = ci[2 * p + 1];
        v8h a0 = *(const v8h*)(xa + c0 * 16);               // ds_load_b128
        v8h a1 = *(const v8h*)(xa + c1 * 16);
        const _Float16* wp = wr + (size_t)(2 * p) * 256 + nm * 16 + 8 * half;
        v8h b0 = *(const v8h*)(wp);                         // global_load_b128
        v8h b1 = *(const v8h*)(wp + 256);
        v16h A = __builtin_shufflevector(a0, a1, 0,1,2,3,4,5,6,7,8,9,10,11,12,13,14,15);
        v16h B = __builtin_shufflevector(b0, b1, 0,1,2,3,4,5,6,7,8,9,10,11,12,13,14,15);
        acc0 = __builtin_amdgcn_wmma_f32_16x16x32_f16(false, A, false, B,
                                                      (short)0, acc0, false, false);
      }
      // ---- pair p+1: k-blocks (2p+2, 2p+3) -> acc1
      {
        const int c0 = ci[2 * p + 2];
        const int c1 = ci[2 * p + 3];
        v8h a0 = *(const v8h*)(xa + c0 * 16);
        v8h a1 = *(const v8h*)(xa + c1 * 16);
        const _Float16* wp = wr + (size_t)(2 * p + 2) * 256 + nm * 16 + 8 * half;
        v8h b0 = *(const v8h*)(wp);
        v8h b1 = *(const v8h*)(wp + 256);
        v16h A = __builtin_shufflevector(a0, a1, 0,1,2,3,4,5,6,7,8,9,10,11,12,13,14,15);
        v16h B = __builtin_shufflevector(b0, b1, 0,1,2,3,4,5,6,7,8,9,10,11,12,13,14,15);
        acc1 = __builtin_amdgcn_wmma_f32_16x16x32_f16(false, A, false, B,
                                                      (short)0, acc1, false, false);
      }
    }

    // ---- Store 16x16 f32 tile: lane (n, half), VGPR j -> row m = 8*half + j
    float* op = out + (size_t)(row_base + mgroup * 16 + 8 * half) * OUT_DIM
                    + r * 16 + nm;
#pragma unroll
    for (int j = 0; j < 8; ++j) {
      op[(size_t)j * OUT_DIM] = acc0[j] + acc1[j];
    }
  }
}

// ---------------------------------------------------------------------------
extern "C" void kernel_launch(void* const* d_in, const int* in_sizes, int n_in,
                              void* d_out, int out_size, void* d_ws, size_t ws_size,
                              hipStream_t stream) {
  (void)in_sizes; (void)n_in; (void)out_size; (void)ws_size;

  const float* x   = (const float*)d_in[0];  // (4,2048,4096) fp32
  const float* w   = (const float*)d_in[1];  // (256,128,16,16) fp32
  const int*   ci  = (const int*)d_in[2];    // (256,128) int32
  float*       out = (float*)d_out;          // (4,2048,4096) fp32
  _Float16*    wh  = (_Float16*)d_ws;        // 16.8 MB f16 weights

  // Convert weights fp32 -> f16 (idempotent, runs every call).
  wcvt_kernel<<<(R_DIM * K_DIM * 256 / 4) / 256, 256, 0, stream>>>(
      (const v4f*)w, (v4h*)wh);

  // Block-sparse GEMM.
  bsl_kernel<<<M_TOTAL / M_TILE, THREADS, 0, stream>>>(x, wh, ci, out);
}
